// WindowsAttention_21852793602092
// MI455X (gfx1250) — matrix operand order repaired and checked by reading, not compiled, not run on hardware
//
#include <hip/hip_runtime.h>
#include <hip/hip_bf16.h>

typedef __attribute__((ext_vector_type(16))) __bf16 v16bf;
typedef __attribute__((ext_vector_type(8)))  float  v8f;

// ---------- helpers ----------
__device__ __forceinline__ unsigned short f2bf(float f) {
    unsigned u = __float_as_uint(f);
    u = (u + 0x7FFFu + ((u >> 16) & 1u)) >> 16;   // round-to-nearest-even
    return (unsigned short)u;
}
__device__ __forceinline__ unsigned f2bf2(float lo, float hi) {
    return (unsigned)f2bf(lo) | ((unsigned)f2bf(hi) << 16);
}
__device__ __forceinline__ float bf2f(unsigned short b) {
    return __uint_as_float(((unsigned)b) << 16);
}

// A fragment: 16x32 bf16, rows m0..m0+15, K window [k0,k0+32) from row-major src
__device__ __forceinline__ v16bf load_a(const unsigned short* base, int stride,
                                        int m0, int k0, int lane) {
    union { v16bf v; unsigned u[8]; } f;
    const int m = m0 + (lane & 15);
    const int half = lane >> 4;
    const unsigned short* row = base + m * stride;
#pragma unroll
    for (int r = 0; r < 8; ++r) {
        int kk = k0 + ((r >> 2) << 4) + (half << 3) + ((r & 3) << 1);
        f.u[r] = *reinterpret_cast<const unsigned*>(row + kk);
    }
    return f.v;
}

// B fragment: B[k][n] = src[n*stride + k]  (src row n holds K contiguously)
__device__ __forceinline__ v16bf load_bt(const unsigned short* base, int stride,
                                         int n0, int k0, int lane) {
    union { v16bf v; unsigned u[8]; } f;
    const int n = n0 + (lane & 15);
    const int half = lane >> 4;
    const unsigned short* row = base + n * stride;
#pragma unroll
    for (int r = 0; r < 8; ++r) {
        int kk = k0 + ((r >> 2) << 4) + (half << 3) + ((r & 3) << 1);
        f.u[r] = *reinterpret_cast<const unsigned*>(row + kk);
    }
    return f.v;
}

__device__ __forceinline__ v8f wmma_bf16(v16bf a, v16bf b, v8f c) {
    return __builtin_amdgcn_wmma_f32_16x16x32_bf16(false, a, false, b,
                                                   (short)0, c, false, false);
}

// ---------- prep kernels ----------
__global__ void cvt_bf16_kernel(const float* __restrict__ src,
                                unsigned short* __restrict__ dst, int n) {
    int i = blockIdx.x * blockDim.x + threadIdx.x;
    if (i < n) dst[i] = f2bf(src[i]);
}

__global__ void build_bias_kernel(const int* __restrict__ rpi,
                                  const float* __restrict__ table,
                                  float* __restrict__ bias) {
    int i = blockIdx.x * blockDim.x + threadIdx.x;   // H*64*64 = 24576
    if (i >= 6 * 64 * 64) return;
    int h = i / 4096;
    int ij = i % 4096;
    bias[i] = table[rpi[ij] * 6 + h];
}

// ---------- fused per-window attention ----------
__global__ __launch_bounds__(256) void fused_window_attn(
    const float* __restrict__ x,          // [8192,64,192]
    const float* __restrict__ attn_mask,  // [1024,64,64]
    const float* __restrict__ qkv_b,      // [576]
    const float* __restrict__ proj_b,     // [192]
    const float* __restrict__ gating_b,   // [192]
    const unsigned short* __restrict__ wq,  // bf16 [576,192]
    const unsigned short* __restrict__ wg,  // bf16 [192,192]
    const unsigned short* __restrict__ wp,  // bf16 [192,192]
    const float* __restrict__ biasb,        // [6,64,64]
    float* __restrict__ out)                // [8192,64,192]
{
    __shared__ unsigned short xs[64 * 192];      // x bf16; reused as gated-attn-out
    __shared__ unsigned short qs[6 * 64 * 32];   // q (pre-scaled) bf16
    __shared__ unsigned short ksm[6 * 64 * 32];  // k bf16
    __shared__ unsigned short vts[6 * 32 * 64];  // v transposed [h][d][tok] bf16
    __shared__ unsigned short gsig[64 * 192];    // sigmoid(gate) bf16
    __shared__ float          Sbuf[64 * 64];     // scores f32 (one head)
    __shared__ unsigned short Pbuf[64 * 64];     // softmax probs bf16 (one head)

    const int b      = blockIdx.x;
    const int tid    = threadIdx.x;
    const int lane   = tid & 31;
    const int wid    = tid >> 5;
    const int half   = lane >> 4;
    const int lane15 = lane & 15;
    const int w      = b & 1023;                 // window-mask index = b % NW

    // ---- phase 1: x -> bf16 LDS ----
    const float* xg = x + (size_t)b * (64 * 192);
    unsigned* xs32 = reinterpret_cast<unsigned*>(xs);
    for (int i = tid; i < 64 * 192 / 4; i += 256) {
        float4 v4 = reinterpret_cast<const float4*>(xg)[i];
        xs32[2 * i]     = f2bf2(v4.x, v4.y);
        xs32[2 * i + 1] = f2bf2(v4.z, v4.w);
    }
    __syncthreads();

    const float scale = 0.17677669529663687f;    // 32^-0.5

    // ---- phase 2: QKV + gating GEMM (M=64, Nout=768, K=192) ----
    // Wave keeps one column-tile nt, hoists the 6 global B fragments,
    // sweeps all 4 row-tiles. Region routing is wave-uniform (scalar branches).
    for (int i = 0; i < 6; ++i) {
        const int nt  = wid + i * 8;                      // 0..47
        const int n0u = __builtin_amdgcn_readfirstlane(nt * 16);
        const bool isqkv = (n0u < 576);
        const unsigned short* wsrc = isqkv ? wq : wg;
        const int   nrel = isqkv ? n0u : n0u - 576;
        const float bv   = (isqkv ? qkv_b : gating_b)[nrel + lane15];

        v16bf Bf[6];
#pragma unroll
        for (int ks = 0; ks < 6; ++ks)
            Bf[ks] = load_bt(wsrc, 192, nrel, ks * 32, lane);

#pragma unroll
        for (int mt = 0; mt < 4; ++mt) {
            v8f acc = {0.f, 0.f, 0.f, 0.f, 0.f, 0.f, 0.f, 0.f};
#pragma unroll
            for (int ks = 0; ks < 6; ++ks)
                acc = wmma_bf16(load_a(xs, 192, mt * 16, ks * 32, lane), Bf[ks], acc);

            if (n0u < 192) {                                // q (scaled)
                const int h = n0u >> 5, d = (n0u + lane15) & 31;
#pragma unroll
                for (int r = 0; r < 8; ++r) {
                    const int m = mt * 16 + r + (half << 3);
                    qs[(h * 64 + m) * 32 + d] = f2bf((acc[r] + bv) * scale);
                }
            } else if (n0u < 384) {                         // k
                const int c2 = n0u - 192, h = c2 >> 5, d = (c2 + lane15) & 31;
#pragma unroll
                for (int r = 0; r < 8; ++r) {
                    const int m = mt * 16 + r + (half << 3);
                    ksm[(h * 64 + m) * 32 + d] = f2bf(acc[r] + bv);
                }
            } else if (n0u < 576) {                         // v (store transposed)
                const int c2 = n0u - 384, h = c2 >> 5, d = (c2 + lane15) & 31;
#pragma unroll
                for (int r = 0; r < 8; ++r) {
                    const int m = mt * 16 + r + (half << 3);
                    vts[(h * 32 + d) * 64 + m] = f2bf(acc[r] + bv);
                }
            } else {                                        // gate -> sigmoid
                const int c2 = n0u - 576 + lane15;
#pragma unroll
                for (int r = 0; r < 8; ++r) {
                    const int m = mt * 16 + r + (half << 3);
                    const float g = acc[r] + bv;
                    gsig[m * 192 + c2] =
                        f2bf(__builtin_amdgcn_rcpf(1.f + __expf(-g)));
                }
            }
        }
    }
    __syncthreads();

    // ---- phase 3: attention per head ----
    const float* maskp = attn_mask + (size_t)w * 4096;
    for (int h = 0; h < 6; ++h) {
        // S = q @ k^T : wave owns nt = wid&3, rows mt = (wid>>2) and (wid>>2)+2;
        // shared B fragment hoisted across both tiles. K=32 = single WMMA.
        {
            const int ntS = wid & 3, mt0 = wid >> 2;
            v16bf bS = load_bt(ksm + h * 2048, 32, ntS * 16, 0, lane);
            const int n = ntS * 16 + lane15;
#pragma unroll
            for (int j = 0; j < 2; ++j) {
                const int mt = mt0 + j * 2;
                v8f acc = {0.f, 0.f, 0.f, 0.f, 0.f, 0.f, 0.f, 0.f};
                acc = wmma_bf16(load_a(qs + h * 2048, 32, mt * 16, 0, lane), bS, acc);
#pragma unroll
                for (int r = 0; r < 8; ++r)
                    Sbuf[(mt * 16 + r + (half << 3)) * 64 + n] = acc[r];
            }
        }
        __syncthreads();

        // softmax rows (bias + mask fused); wave handles 8 rows, 2 cols/lane
        const float* bh = biasb + h * 4096;
#pragma unroll
        for (int i = 0; i < 8; ++i) {
            const int row = wid * 8 + i;
            const int o0 = row * 64 + lane;
            float a0 = Sbuf[o0]      + bh[o0]      + maskp[o0];
            float a1 = Sbuf[o0 + 32] + bh[o0 + 32] + maskp[o0 + 32];
            float mx = fmaxf(a0, a1);
#pragma unroll
            for (int o = 16; o > 0; o >>= 1) mx = fmaxf(mx, __shfl_xor(mx, o, 32));
            float e0 = __expf(a0 - mx), e1 = __expf(a1 - mx);
            float sm = e0 + e1;
#pragma unroll
            for (int o = 16; o > 0; o >>= 1) sm += __shfl_xor(sm, o, 32);
            const float inv = __builtin_amdgcn_rcpf(sm);
            Pbuf[o0]      = f2bf(e0 * inv);
            Pbuf[o0 + 32] = f2bf(e1 * inv);
        }
        __syncthreads();

        // O = P @ v : 4x2 tiles, one per wave; fuse sigmoid gating, store bf16
        {
            const int mt = wid >> 1, nt = wid & 1;
            v8f acc = {0.f, 0.f, 0.f, 0.f, 0.f, 0.f, 0.f, 0.f};
#pragma unroll
            for (int ks = 0; ks < 2; ++ks) {
                v16bf a  = load_a(Pbuf, 64, mt * 16, ks * 32, lane);
                v16bf bb = load_bt(vts + h * 2048, 64, nt * 16, ks * 32, lane);
                acc = wmma_bf16(a, bb, acc);
            }
            const int col = h * 32 + nt * 16 + lane15;
#pragma unroll
            for (int r = 0; r < 8; ++r) {
                const int m = mt * 16 + r + (half << 3);
                const float g = bf2f(gsig[m * 192 + col]);
                xs[m * 192 + col] = f2bf(acc[r] * g);   // reuse xs as y
            }
        }
    }
    __syncthreads();

    // ---- phase 4: proj GEMM (M=64, N=192, K=192) + bias -> global f32 ----
    // Wave keeps a fixed row-tile mt = wid&3, hoists 6 A fragments from LDS,
    // sweeps nt = (wid>>2) + 2i  (6 column tiles per wave).
    {
        const int mt = wid & 3;
        v16bf Af[6];
#pragma unroll
        for (int ks = 0; ks < 6; ++ks)
            Af[ks] = load_a(xs, 192, mt * 16, ks * 32, lane);

        float* og = out + (size_t)b * (64 * 192);
        for (int i = 0; i < 6; ++i) {
            const int nt = (wid >> 2) + 2 * i;            // 0..11
            v8f acc = {0.f, 0.f, 0.f, 0.f, 0.f, 0.f, 0.f, 0.f};
#pragma unroll
            for (int ks = 0; ks < 6; ++ks)
                acc = wmma_bf16(Af[ks], load_bt(wp, 192, nt * 16, ks * 32, lane), acc);

            const int n = nt * 16 + lane15;
            const float pb = proj_b[n];
#pragma unroll
            for (int r = 0; r < 8; ++r) {
                const int m = mt * 16 + r + (half << 3);
                og[m * 192 + n] = acc[r] + pb;
            }
        }
    }
}

// ---------- launcher ----------
extern "C" void kernel_launch(void* const* d_in, const int* in_sizes, int n_in,
                              void* d_out, int out_size, void* d_ws, size_t ws_size,
                              hipStream_t stream) {
    const float* x        = (const float*)d_in[0];
    const int*   rpi      = (const int*)d_in[1];
    const float* attn_mask= (const float*)d_in[2];
    const float* qkv_w    = (const float*)d_in[3];
    const float* qkv_b    = (const float*)d_in[4];
    const float* proj_w   = (const float*)d_in[5];
    const float* proj_b   = (const float*)d_in[6];
    const float* gating_w = (const float*)d_in[7];
    const float* gating_b = (const float*)d_in[8];
    const float* rpb      = (const float*)d_in[9];
    float* out = (float*)d_out;

    char* ws = (char*)d_ws;
    unsigned short* wq = (unsigned short*)(ws);              // 576*192
    unsigned short* wg = (unsigned short*)(ws + 221184);     // 192*192
    unsigned short* wp = (unsigned short*)(ws + 294912);     // 192*192
    float* biasb       = (float*)(ws + 368640);              // 6*64*64 f32

    cvt_bf16_kernel<<<(110592 + 255) / 256, 256, 0, stream>>>(qkv_w, wq, 110592);
    cvt_bf16_kernel<<<(36864 + 255) / 256, 256, 0, stream>>>(gating_w, wg, 36864);
    cvt_bf16_kernel<<<(36864 + 255) / 256, 256, 0, stream>>>(proj_w, wp, 36864);
    build_bias_kernel<<<96, 256, 0, stream>>>(rpi, rpb, biasb);

    fused_window_attn<<<8192, 256, 0, stream>>>(
        x, attn_mask, qkv_b, proj_b, gating_b, wq, wg, wp, biasb, out);
}
